// NeRFNetworkPartial_21930103013787
// MI455X (gfx1250) — matrix operand order repaired and checked by reading, not compiled, not run on hardware
//
#include <hip/hip_runtime.h>
#include <math.h>
#include <stdint.h>

typedef _Float16 v16h __attribute__((ext_vector_type(16)));
typedef _Float16 v8h  __attribute__((ext_vector_type(8)));
typedef _Float16 v2h  __attribute__((ext_vector_type(2)));
typedef float    v8f  __attribute__((ext_vector_type(8)));
typedef uint32_t v4u  __attribute__((ext_vector_type(4)));
typedef uint32_t v8u  __attribute__((ext_vector_type(8)));

#define N_LEVELS 20
#define LOG2_T   24
#define TBL_MASK ((1u << LOG2_T) - 1u)
#define FEAT_PAD 64          // 40 real features zero-padded to 64 for WMMA K-steps
#define WAVES_PER_BLOCK 8

struct ResArr { float r[N_LEVELS]; };

// ---------------------------------------------------------------------------
// Kernel 1: instant-NGP hash-grid encode.  Gather-bound: 160 random 4B loads
// per point from the 1.34GB table.  Unroll levels x2 => 16 gathers in flight
// per thread.  Packed fp16 accumulate (v_pk_fma_f16) to match the reference.
// Writes [N, 64] fp16 rows (cols 40..63 zeroed) for direct WMMA consumption.
// ---------------------------------------------------------------------------
__global__ __launch_bounds__(256) void hashgrid_encode_kernel(
    const float* __restrict__ x, const uint32_t* __restrict__ table,
    _Float16* __restrict__ feat, ResArr res, int npts)
{
  int i = blockIdx.x * blockDim.x + threadIdx.x;
  if (i >= npts) return;

  float px = x[3 * i + 0];
  float py = x[3 * i + 1];
  float pz = x[3 * i + 2];
  _Float16* frow = feat + (size_t)i * FEAT_PAD;

#pragma unroll 2
  for (int l = 0; l < N_LEVELS; ++l) {
    float r  = res.r[l];
    float fx = px * r, fy = py * r, fz = pz * r;
    float bx = floorf(fx), by = floorf(fy), bz = floorf(fz);
    float tx = fx - bx, ty = fy - by, tz = fz - bz;
    uint32_t cx = (uint32_t)(int)bx;
    uint32_t cy = (uint32_t)(int)by;
    uint32_t cz = (uint32_t)(int)bz;
    const uint32_t* tl = table + ((size_t)l << LOG2_T);  // one u32 == 2 fp16 feats

    v2h acc = {};
#pragma unroll
    for (int corner = 0; corner < 8; ++corner) {
      uint32_t ox = corner & 1u, oy = (corner >> 1) & 1u, oz = (corner >> 2) & 1u;
      uint32_t idx = ((cx + ox) * 1u
                    ^ (cy + oy) * 2654435761u
                    ^ (cz + oz) * 805459861u) & TBL_MASK;
      float w = (ox ? tx : 1.0f - tx) * (oy ? ty : 1.0f - ty) * (oz ? tz : 1.0f - tz);
      union { uint32_t u; v2h v; } cvt;
      cvt.u = tl[idx];
      _Float16 wh = (_Float16)w;
      v2h wv = {wh, wh};
      acc += wv * cvt.v;                 // v_pk_fma_f16
    }
    union { uint32_t u; v2h v; } o;
    o.v = acc;
    *(uint32_t*)(frow + 2 * l) = o.u;    // 4B-aligned packed store
  }
  // zero the K=40..63 padding (16B-aligned chunks: byte offsets 80/96/112)
  v8h z = {};
  *(v8h*)(frow + 40) = z;
  *(v8h*)(frow + 48) = z;
  *(v8h*)(frow + 56) = z;
}

// ---------------------------------------------------------------------------
// TDM helpers: build a D# (groups 0+1) for a 2D fp16 tile and issue
// tensor_load_to_lds.  tile dims larger than tensor dims give zero fill (OOB
// reads return 0), which we exploit for the K 40->64 / N 4->16 padding.
// ---------------------------------------------------------------------------
static __device__ __forceinline__ uint32_t lds_off_u32(const void* p) {
  // generic -> LDS addrspace; ptrtoint of AS(3) pointer == 32-bit LDS byte offset
  return (uint32_t)(uintptr_t)(__attribute__((address_space(3))) const void*)p;
}

static __device__ __forceinline__ void tdm_load_2d_f16(
    const void* gsrc, uint32_t lds_byte_off,
    uint32_t dim0, uint32_t dim1,      // tensor dims (fp16 elements)
    uint32_t tile0, uint32_t tile1,    // tile dims   (fp16 elements)
    uint32_t stride0)                  // row stride  (fp16 elements)
{
  uint64_t ga = (uint64_t)(uintptr_t)gsrc;
  v4u g0;
  g0[0] = 1u;                                              // count=1, load desc
  g0[1] = lds_byte_off;                                    // lds_addr
  g0[2] = (uint32_t)ga;                                    // global_addr[31:0]
  g0[3] = (uint32_t)((ga >> 32) & 0x01FFFFFFu) | (2u << 30); // addr[56:32] | type=2
  v8u g1;
  g1[0] = 1u << 16;                                        // data_size=1 -> 2 bytes
  g1[1] = (dim0 & 0xFFFFu) << 16;                          // tensor_dim0[15:0] @63:48
  g1[2] = (dim0 >> 16) | ((dim1 & 0xFFFFu) << 16);         // dim0 hi | dim1 lo
  g1[3] = (dim1 >> 16) | ((tile0 & 0xFFFFu) << 16);        // dim1 hi | tile_dim0
  g1[4] = (tile1 & 0xFFFFu);                               // tile_dim1 (tile_dim2=0)
  g1[5] = stride0;                                         // tensor_dim0_stride lo32
  g1[6] = 0u;
  g1[7] = 0u;
  asm volatile("tensor_load_to_lds %0, %1" : : "s"(g0), "s"(g1) : "memory");
}

// ---------------------------------------------------------------------------
// Kernel 2: fused MLP via v_wmma_f32_16x16x32_f16.  One wave == 16 points.
// Weights staged by the Tensor Data Mover (async DMA) with OOB zero-padding,
// then transposed in LDS for column-contiguous B-fragment loads.
// Layer1: [16,64(pad40)] x [64(pad),64] = 4 N-tiles x 2 K-steps = 8 WMMAs.
// ReLU -> LDS -> Layer2: [16,64] x [64,16(pad4)] = 2 WMMAs.
// ---------------------------------------------------------------------------
static __device__ __forceinline__ v16h hcat(v8h lo, v8h hi) {
  return __builtin_shufflevector(lo, hi, 0, 1, 2, 3, 4, 5, 6, 7,
                                         8, 9, 10, 11, 12, 13, 14, 15);
}
static __device__ __forceinline__ v8f wmma16(v16h a, v16h b, v8f c) {
  return __builtin_amdgcn_wmma_f32_16x16x32_f16(
      /*neg_a=*/false, a, /*neg_b=*/false, b,
      /*c_mod=*/(short)0, c, /*reuse_a=*/false, /*reuse_b=*/false);
}

__global__ __launch_bounds__(256) void hashmlp_wmma_kernel(
    const _Float16* __restrict__ feat, const _Float16* __restrict__ W1,
    const _Float16* __restrict__ W2, float* __restrict__ out, int npts)
{
  __shared__ alignas(16) _Float16 w1s[64 * 64];  // TDM staging, row-major [k][n], k>=40 zero
  __shared__ alignas(16) _Float16 w2s[64 * 16];  // TDM staging, row-major [k][n], n>=4 zero
  __shared__ alignas(16) _Float16 w1t[64 * 64];  // transposed [n][k]
  __shared__ alignas(16) _Float16 w2t[16 * 64];  // transposed [n][k]
  __shared__ alignas(16) _Float16 hbuf[WAVES_PER_BLOCK][16 * 64];

  int tid = threadIdx.x;

  // --- async weight staging: wave 0 drives the TDM (EXEC ignored by TDM) ---
  if (tid < 32) {
    // W1 [40,64] -> w1s[64][64]: tile rows 40..63 OOB => zero K-padding
    tdm_load_2d_f16(W1, lds_off_u32(w1s), /*dim0=*/64, /*dim1=*/40,
                    /*tile0=*/64, /*tile1=*/64, /*stride0=*/64);
    // W2 [64,4] -> w2s[64][16]: tile cols 4..15 OOB => zero N-padding
    tdm_load_2d_f16(W2, lds_off_u32(w2s), /*dim0=*/4, /*dim1=*/64,
                    /*tile0=*/16, /*tile1=*/64, /*stride0=*/4);
    __builtin_amdgcn_s_wait_tensorcnt(0);
  }
  __syncthreads();

  // --- transpose staging -> [n][k] for contiguous-K B-fragment loads ---
  for (int s = tid; s < 64 * 64; s += 256) {
    int n = s >> 6, k = s & 63;
    w1t[s] = w1s[k * 64 + n];
  }
  for (int s = tid; s < 16 * 64; s += 256) {
    int n = s >> 6, k = s & 63;
    w2t[s] = w2s[k * 16 + n];
  }
  __syncthreads();

  int wave = tid >> 5;
  int lane = tid & 31;
  int m    = lane & 15;      // row (A) or column (B/D) index within tile
  int hi   = lane >> 4;      // lane half select
  int row0 = (blockIdx.x * WAVES_PER_BLOCK + wave) * 16;
  bool active = (row0 + 16) <= npts;   // wave-uniform: EXEC stays all-ones

  // A fragment (16-bit A 16x32 layout): halves0..7 = K[hi*8..+7], halves8..15 = K[16+hi*8..+7]
  int ka = hi * 8;
  // B fragment (16-bit B 32x16 layout): halves0..7 = K[hi*8..+7], halves8..15 = K[16+hi*8..+7] of one column
  int kb = hi * 8;

  v16h a0 = {}, a1 = {};
  if (active) {
    const _Float16* arow = feat + (size_t)(row0 + m) * FEAT_PAD;
    a0 = hcat(*(const v8h*)(arow + ka),      *(const v8h*)(arow + ka + 16));
    a1 = hcat(*(const v8h*)(arow + 32 + ka), *(const v8h*)(arow + 32 + ka + 16));
  }

  // ---- Layer 1: 4 output tiles of 16 cols, K = 64 in two 32-steps ----
  v8f c[4];
#pragma unroll
  for (int nt = 0; nt < 4; ++nt) {
    const _Float16* bcol = w1t + ((nt * 16 + m) << 6);
    v16h b0 = hcat(*(const v8h*)(bcol + kb),      *(const v8h*)(bcol + kb + 16));
    v16h b1 = hcat(*(const v8h*)(bcol + 32 + kb), *(const v8h*)(bcol + 32 + kb + 16));
    v8f acc = {};
    acc = wmma16(a0, b0, acc);
    acc = wmma16(a1, b1, acc);
    c[nt] = acc;
  }

  // ---- ReLU, stage h[16][64] fp16 in LDS (C/D layout: row hi*8+j, col nt*16+m) ----
#pragma unroll
  for (int nt = 0; nt < 4; ++nt) {
#pragma unroll
    for (int j = 0; j < 8; ++j) {
      float v = c[nt][j];
      v = v > 0.0f ? v : 0.0f;
      hbuf[wave][(hi * 8 + j) * 64 + nt * 16 + m] = (_Float16)v;
    }
  }
  __syncthreads();

  // ---- Layer 2: [16,64] x [64,16]  (only cols 0..3 real) ----
  const _Float16* hrow = &hbuf[wave][m * 64];
  v16h a20 = hcat(*(const v8h*)(hrow + ka),      *(const v8h*)(hrow + ka + 16));
  v16h a21 = hcat(*(const v8h*)(hrow + 32 + ka), *(const v8h*)(hrow + 32 + ka + 16));
  const _Float16* b2col = w2t + (m << 6);
  v16h b20 = hcat(*(const v8h*)(b2col + kb),      *(const v8h*)(b2col + kb + 16));
  v16h b21 = hcat(*(const v8h*)(b2col + 32 + kb), *(const v8h*)(b2col + 32 + kb + 16));
  v8f d = {};
  d = wmma16(a20, b20, d);
  d = wmma16(a21, b21, d);

  if (active && m < 4) {
#pragma unroll
    for (int j = 0; j < 8; ++j) {
      out[(size_t)(row0 + hi * 8 + j) * 4 + m] = d[j];
    }
  }
}

// ---------------------------------------------------------------------------
extern "C" void kernel_launch(void* const* d_in, const int* in_sizes, int n_in,
                              void* d_out, int out_size, void* d_ws, size_t ws_size,
                              hipStream_t stream)
{
  const float*     x     = (const float*)d_in[0];
  const uint32_t*  table = (const uint32_t*)d_in[1];   // [20][2^24] packed fp16 pairs
  const _Float16*  W1    = (const _Float16*)d_in[2];   // [40,64]
  const _Float16*  W2    = (const _Float16*)d_in[3];   // [64,4]
  float*           out   = (float*)d_out;              // [N,4]
  int npts = in_sizes[0] / 3;

  _Float16* feat = (_Float16*)d_ws;                    // [N,64] fp16 = 64MB scratch

  ResArr res;
  double s = 1.0;
  for (int l = 0; l < N_LEVELS; ++l) {
    res.r[l] = (float)floor(16.0 * s);                 // floor(BASE_RES * 1.3819^l)
    s *= 1.3819;
  }

  int eb = (npts + 255) / 256;
  hashgrid_encode_kernel<<<eb, 256, 0, stream>>>(x, table, feat, res, npts);

  int tiles = (npts + 15) / 16;
  int mb = (tiles + WAVES_PER_BLOCK - 1) / WAVES_PER_BLOCK;
  hashmlp_wmma_kernel<<<mb, 256, 0, stream>>>(feat, W1, W2, out, npts);
}
